// AdaptiveGCN_59545426592086
// MI455X (gfx1250) — compile-verified, matrix operand order checked by reading, not run on hardware
//
#include <hip/hip_runtime.h>
#include <hip/hip_bf16.h>
#include <math.h>

typedef __bf16 bf16;
typedef __attribute__((ext_vector_type(16))) __bf16 v16bf;
typedef __attribute__((ext_vector_type(8)))  __bf16 bf16x8;
typedef __attribute__((ext_vector_type(8)))  float   v8f;
typedef __attribute__((ext_vector_type(4)))  unsigned int uint4v;
typedef __attribute__((ext_vector_type(4)))  int i4v;

#define N_  64
#define C_  64
#define T_  300
#define V_  25
#define O_  64
#define S_  3
#define IN_ 16
#define K_  9
#define VP  32               // V padded to 32 for WMMA
#define XTILE (V_ * C_)      // 1600 bf16 = 3200 B per time-step tile

#if __has_builtin(__builtin_amdgcn_global_load_async_to_lds_b128)
#define HAVE_ASYNC_LDS 1
typedef __attribute__((address_space(1))) i4v* gv4p;   // global int4*
typedef __attribute__((address_space(3))) i4v* lv4p;   // LDS int4*
#else
#define HAVE_ASYNC_LDS 0
#endif

// ---------------- WMMA helpers ----------------

static __device__ __forceinline__ v8f wmma_bf16(v16bf a, v16bf b, v8f c) {
  // D(16x16,f32) = A(16x32,bf16) * B(32x16,bf16) + C
  return __builtin_amdgcn_wmma_f32_16x16x32_bf16(false, a, false, b, (short)0, c,
                                                 false, false);
}

// Load one 16x32 bf16 fragment row-chunk for this lane.
// p must point at &row_base[8*sel] where sel = lane>>4; second chunk at +16.
// Layout (ISA 7.12.2, 16-bit A 16x32): lanes0-15 elems0..7=K0..7, elems8..15=K16..23;
// lanes16-31 elems0..7=K8..15, elems8..15=K24..31.  B uses the symmetric layout.
static __device__ __forceinline__ v16bf ldfrag(const bf16* p) {
  bf16x8 lo = *(const bf16x8*)p;
  bf16x8 hi = *(const bf16x8*)(p + 16);
  v16bf r;
#pragma unroll
  for (int j = 0; j < 8; ++j) { r[j] = lo[j]; r[j + 8] = hi[j]; }
  return r;
}

static __device__ __forceinline__ v16bf zfrag() {
  v16bf r;
#pragma unroll
  for (int j = 0; j < 16; ++j) r[j] = (bf16)0.0f;
  return r;
}

// ---------------- async x-tile staging ----------------
// Stage one 3200B x-tile (xT[n][t][*][*]) into LDS.  Threads 0..99 each move
// two 16B chunks so per-thread ASYNCcnt contribution is uniform (=2 per tile).
static __device__ __forceinline__ void stage_tile(const bf16* __restrict__ gsrc,
                                                  bf16* __restrict__ ldst, int tid) {
  if (tid < 100) {
#if HAVE_ASYNC_LDS
    __builtin_amdgcn_global_load_async_to_lds_b128(
        (gv4p)(gsrc + tid * 8), (lv4p)(ldst + tid * 8), 0, 0);
    __builtin_amdgcn_global_load_async_to_lds_b128(
        (gv4p)(gsrc + (tid + 100) * 8), (lv4p)(ldst + (tid + 100) * 8), 0, 0);
#else
    *(uint4v*)(ldst + tid * 8)         = *(const uint4v*)(gsrc + tid * 8);
    *(uint4v*)(ldst + (tid + 100) * 8) = *(const uint4v*)(gsrc + (tid + 100) * 8);
#endif
  }
}

// Retire the current tile's async loads (leave <=2 in flight for the next tile).
static __device__ __forceinline__ void wait_tile_pipelined() {
#if HAVE_ASYNC_LDS
#if __has_builtin(__builtin_amdgcn_s_wait_asynccnt)
  __builtin_amdgcn_s_wait_asynccnt(2);
#else
  asm volatile("s_wait_asynccnt 0x2" ::: "memory");
#endif
#endif
}

static __device__ __forceinline__ void wait_tile_all() {
#if HAVE_ASYNC_LDS
#if __has_builtin(__builtin_amdgcn_s_wait_asynccnt)
  __builtin_amdgcn_s_wait_asynccnt(0);
#else
  asm volatile("s_wait_asynccnt 0x0" ::: "memory");
#endif
#endif
}

// ---------------- prep kernels ----------------

// xT[n][t][v][c] (bf16)  <-  x[n][c][t][v] (f32)
__global__ void agcn_prep_xT(const float* __restrict__ x, bf16* __restrict__ xT) {
  int idx = blockIdx.x * blockDim.x + threadIdx.x;          // over N*T*V
  if (idx >= N_ * T_ * V_) return;
  int v = idx % V_, t = (idx / V_) % T_, n = idx / (V_ * T_);
  const float* xp = x + ((size_t)n * C_ * T_ * V_) + (size_t)t * V_ + v;
  bf16* o = xT + (size_t)idx * C_;
#pragma unroll 8
  for (int c = 0; c < C_; ++c) o[c] = (bf16)xp[(size_t)c * T_ * V_];
}

__global__ void agcn_prep_w(const float* __restrict__ wa, const float* __restrict__ wb,
                            const float* __restrict__ wd, bf16* __restrict__ wah,
                            bf16* __restrict__ wbh, bf16* __restrict__ wdh) {
  int i = blockIdx.x * blockDim.x + threadIdx.x;
  if (i < S_ * O_ * C_) {
    wah[i] = (bf16)wa[i]; wbh[i] = (bf16)wb[i]; wdh[i] = (bf16)wd[i];
  }
}

// se_in[n][c][t] = mean_v x[n][c][t][v]
__global__ void agcn_se_mean(const float* __restrict__ x, float* __restrict__ se_in) {
  int idx = blockIdx.x * blockDim.x + threadIdx.x;          // over N*C*T
  if (idx >= N_ * C_ * T_) return;
  const float* p = x + (size_t)idx * V_;
  float s = 0.f;
#pragma unroll
  for (int v = 0; v < V_; ++v) s += p[v];
  se_in[idx] = s * (1.0f / V_);
}

// se1[i][n][j][t] = relu(conv1d(se_in, w1[i])), 'same' pad 4
__global__ void agcn_se_conv1(const float* __restrict__ se_in, const float* __restrict__ w1,
                              const float* __restrict__ b1, float* __restrict__ se1) {
  int idx = blockIdx.x * blockDim.x + threadIdx.x;          // over S*N*IN*T
  if (idx >= S_ * N_ * IN_ * T_) return;
  int t = idx % T_, j = (idx / T_) % IN_, n = (idx / (T_ * IN_)) % N_,
      i = idx / (T_ * IN_ * N_);
  const float* w  = w1 + ((size_t)i * IN_ + j) * C_ * K_;
  const float* sp = se_in + (size_t)n * C_ * T_;
  float acc = b1[i * IN_ + j];
  for (int c = 0; c < C_; ++c) {
    const float* row = sp + c * T_;
    const float* wr  = w + c * K_;
#pragma unroll
    for (int k = 0; k < K_; ++k) {
      int tt = t + k - (K_ - 1) / 2;
      if (tt >= 0 && tt < T_) acc += wr[k] * row[tt];
    }
  }
  se1[idx] = acc > 0.f ? acc : 0.f;
}

// se[i][n][t] = sigmoid(conv1d(se1, w2[i]))
__global__ void agcn_se_conv2(const float* __restrict__ se1, const float* __restrict__ w2,
                              const float* __restrict__ b2, float* __restrict__ se) {
  int idx = blockIdx.x * blockDim.x + threadIdx.x;          // over S*N*T
  if (idx >= S_ * N_ * T_) return;
  int t = idx % T_, n = (idx / T_) % N_, i = idx / (T_ * N_);
  const float* w  = w2 + (size_t)i * IN_ * K_;
  const float* sp = se1 + ((size_t)i * N_ + n) * IN_ * T_;
  float acc = b2[i];
  for (int j = 0; j < IN_; ++j) {
#pragma unroll
    for (int k = 0; k < K_; ++k) {
      int tt = t + k - (K_ - 1) / 2;
      if (tt >= 0 && tt < T_) acc += w[j * K_ + k] * sp[j * T_ + tt];
    }
  }
  se[idx] = 1.0f / (1.0f + __expf(-acc));
}

// ---------------- attention kernel ----------------
// One block per (subset i, batch n).  128 threads = 4 waves.
// x tiles are staged global->LDS with async loads, double buffered, so the
// WMMA loop never waits on global memory.
// Stage 1 (per t): wave w computes a1,a2 rows o=[16w,16w+16) for all u (V padded
// to 32) via WMMA (K=c), +bias, stores bf16 to LDS as [u][o].
// Stage 2: wave w owns logits tile (w>>1, w&1), accumulates K=(o,t)=19200.
// Epilogue: softmax over u, A_eff = PA + alpha*att, stored transposed [v][u] bf16.
__global__ __launch_bounds__(128) void agcn_attn(
    const bf16* __restrict__ xT, const bf16* __restrict__ wah,
    const bf16* __restrict__ wbh, const float* __restrict__ ba,
    const float* __restrict__ bb, const float* __restrict__ PA,
    const float* __restrict__ alpha, bf16* __restrict__ aeffT) {
  __shared__ __align__(16) bf16 xbuf[2][XTILE];
  __shared__ __align__(16) bf16 a1_lds[VP * O_];
  __shared__ __align__(16) bf16 a2_lds[VP * O_];
  __shared__ float logits[VP * VP];

  int bid = blockIdx.x;
  int i = bid / N_, n = bid % N_;
  int tid = threadIdx.x, wave = tid >> 5, lane = tid & 31;
  int r = lane & 15, sel = lane >> 4;

  const bf16* wA = wah + (size_t)i * O_ * C_;
  const bf16* wB = wbh + (size_t)i * O_ * C_;
  const bf16* xn = xT + (size_t)n * T_ * V_ * C_;

  // Preload weight A-fragments (rows o = 16*wave + r) and per-lane biases.
  v16bf fa[2], fbw[2];
  {
    int row = wave * 16 + r;
#pragma unroll
    for (int ks = 0; ks < 2; ++ks) {
      fa[ks]  = ldfrag(wA + row * C_ + ks * 32 + 8 * sel);
      fbw[ks] = ldfrag(wB + row * C_ + ks * 32 + 8 * sel);
    }
  }
  float ba_l[8], bb_l[8];
#pragma unroll
  for (int g = 0; g < 8; ++g) {
    int o = wave * 16 + 8 * sel + g;
    ba_l[g] = ba[i * O_ + o];
    bb_l[g] = bb[i * O_ + o];
  }

  int mu = wave >> 1, nw = wave & 1;
  v8f acc2 = {0.f, 0.f, 0.f, 0.f, 0.f, 0.f, 0.f, 0.f};

  // prologue: stage tile 0
  stage_tile(xn, xbuf[0], tid);

  for (int t = 0; t < T_; ++t) {
    int cur = t & 1;
    // issue next tile, retire current tile
    if (t + 1 < T_) {
      stage_tile(xn + (size_t)(t + 1) * XTILE, xbuf[cur ^ 1], tid);
      wait_tile_pipelined();          // asynccnt <= 2: tile t complete
    } else {
      wait_tile_all();                // asynccnt == 0: last tile complete
    }
    __syncthreads();  // tile t visible to all waves; a1/a2 free for reuse

    const bf16* xt = xbuf[cur];
    // ---- stage 1: a1,a2 tiles -> LDS ----
#pragma unroll
    for (int ut = 0; ut < 2; ++ut) {
      int u = ut * 16 + r;
      v16bf xb0, xb1;
      if (u < V_) {
        xb0 = ldfrag(xt + u * C_ + 8 * sel);
        xb1 = ldfrag(xt + u * C_ + 32 + 8 * sel);
      } else { xb0 = zfrag(); xb1 = zfrag(); }
      v8f d1 = {0.f, 0.f, 0.f, 0.f, 0.f, 0.f, 0.f, 0.f};
      v8f d2 = {0.f, 0.f, 0.f, 0.f, 0.f, 0.f, 0.f, 0.f};
      d1 = wmma_bf16(fa[0], xb0, d1);  d1 = wmma_bf16(fa[1], xb1, d1);
      d2 = wmma_bf16(fbw[0], xb0, d2); d2 = wmma_bf16(fbw[1], xb1, d2);
      bf16x8 s1v, s2v;
#pragma unroll
      for (int g = 0; g < 8; ++g) {
        s1v[g] = (bf16)(d1[g] + ba_l[g]);
        s2v[g] = (bf16)(d2[g] + bb_l[g]);
      }
      int base = u * O_ + wave * 16 + 8 * sel;
      *(bf16x8*)&a1_lds[base] = s1v;
      *(bf16x8*)&a2_lds[base] = s2v;
    }
    __syncthreads();
    // ---- stage 2: logits += a1^T * a2 over K=o (64) ----
#pragma unroll
    for (int ks = 0; ks < 2; ++ks) {
      v16bf af  = ldfrag(&a1_lds[(mu * 16 + r) * O_ + ks * 32 + 8 * sel]);
      v16bf bf2 = ldfrag(&a2_lds[(nw * 16 + r) * O_ + ks * 32 + 8 * sel]);
      acc2 = wmma_bf16(af, bf2, acc2);
    }
  }

  // dump logits tiles to LDS
  __syncthreads();
#pragma unroll
  for (int g = 0; g < 8; ++g)
    logits[(mu * 16 + 8 * sel + g) * VP + nw * 16 + r] = acc2[g];
  __syncthreads();

  // softmax over u (dim -2) per column w; A_eff = PA + alpha*att, store [v][u] bf16
  if (tid < VP) {
    int w = tid;
    bf16* outp = aeffT + (((size_t)i * N_ + n) * VP + w) * VP;
    if (w < V_) {
      const float inv = 1.0f / ((float)O_ * (float)T_);
      float mx = -1e30f;
#pragma unroll
      for (int u = 0; u < V_; ++u) mx = fmaxf(mx, logits[u * VP + w] * inv);
      float ex[V_];
      float ssum = 0.f;
#pragma unroll
      for (int u = 0; u < V_; ++u) {
        ex[u] = __expf(logits[u * VP + w] * inv - mx);
        ssum += ex[u];
      }
      float al = alpha[0], rs = 1.0f / ssum;
#pragma unroll
      for (int u = 0; u < VP; ++u) {
        float val = (u < V_) ? (PA[((size_t)i * V_ + u) * V_ + w] + al * ex[u < V_ ? u : 0] * rs)
                             : 0.0f;
        outp[u] = (bf16)val;
      }
    } else {
#pragma unroll
      for (int u = 0; u < VP; ++u) outp[u] = (bf16)0.0f;
    }
  }
}

// ---------------- output kernel ----------------
// One block per (n, t).  128 threads = 4 waves; wave w owns rows o=[16w,16w+16).
// For each subset: P = Wd*x_t (WMMA, K=c) -> LDS [o][u]; q = P*A_effT (WMMA, K=u=32);
// acc += (1+se)*q + bd.  Writes y once.
__global__ __launch_bounds__(128) void agcn_out(
    const bf16* __restrict__ xT, const bf16* __restrict__ wdh,
    const float* __restrict__ bd, const bf16* __restrict__ aeffT,
    const float* __restrict__ se, float* __restrict__ y) {
  __shared__ __align__(16) bf16 P_lds[O_ * VP];

  int bid = blockIdx.x;
  int n = bid / T_, t = bid % T_;
  int tid = threadIdx.x, wave = tid >> 5, lane = tid & 31;
  int r = lane & 15, sel = lane >> 4;

  const bf16* xt = xT + ((size_t)n * T_ + t) * V_ * C_;

  // preload x B-fragments (shared across subsets): [ut][kstep]
  v16bf xb[2][2];
#pragma unroll
  for (int ut = 0; ut < 2; ++ut) {
    int u = ut * 16 + r;
    if (u < V_) {
      xb[ut][0] = ldfrag(xt + u * C_ + 8 * sel);
      xb[ut][1] = ldfrag(xt + u * C_ + 32 + 8 * sel);
    } else { xb[ut][0] = zfrag(); xb[ut][1] = zfrag(); }
  }

  v8f accv[2];
  accv[0] = (v8f){0.f, 0.f, 0.f, 0.f, 0.f, 0.f, 0.f, 0.f};
  accv[1] = (v8f){0.f, 0.f, 0.f, 0.f, 0.f, 0.f, 0.f, 0.f};

  for (int i = 0; i < S_; ++i) {
    const bf16* wD = wdh + (size_t)i * O_ * C_;
    int row = wave * 16 + r;
    v16bf wf0 = ldfrag(wD + row * C_ + 8 * sel);
    v16bf wf1 = ldfrag(wD + row * C_ + 32 + 8 * sel);
    // stage 1: P tile rows for this wave
#pragma unroll
    for (int ut = 0; ut < 2; ++ut) {
      v8f d = {0.f, 0.f, 0.f, 0.f, 0.f, 0.f, 0.f, 0.f};
      d = wmma_bf16(wf0, xb[ut][0], d);
      d = wmma_bf16(wf1, xb[ut][1], d);
#pragma unroll
      for (int g = 0; g < 8; ++g)
        P_lds[(wave * 16 + 8 * sel + g) * VP + ut * 16 + r] = (bf16)d[g];
    }
    __syncthreads();
    // stage 2: q = P * A_effT, K = u = 32 (one step)
    v16bf pf = ldfrag(&P_lds[(wave * 16 + r) * VP + 8 * sel]);
    const bf16* at = aeffT + ((size_t)i * N_ + n) * VP * VP;
    float gse = 1.0f + se[((size_t)i * N_ + n) * T_ + t];
    float bdl[8];
#pragma unroll
    for (int g = 0; g < 8; ++g) bdl[g] = bd[i * O_ + wave * 16 + 8 * sel + g];
#pragma unroll
    for (int vt = 0; vt < 2; ++vt) {
      v16bf bf2 = ldfrag(at + (vt * 16 + r) * VP + 8 * sel);
      v8f q = {0.f, 0.f, 0.f, 0.f, 0.f, 0.f, 0.f, 0.f};
      q = wmma_bf16(pf, bf2, q);
#pragma unroll
      for (int g = 0; g < 8; ++g) accv[vt][g] += gse * q[g] + bdl[g];
    }
    __syncthreads();
  }

  // write y[n][o][t][v]
#pragma unroll
  for (int vt = 0; vt < 2; ++vt) {
    int v = vt * 16 + r;
    if (v < V_) {
#pragma unroll
      for (int g = 0; g < 8; ++g) {
        int o = wave * 16 + 8 * sel + g;
        y[(((size_t)n * O_ + o) * T_ + t) * V_ + v] = accv[vt][g];
      }
    }
  }
}

// ---------------- launcher ----------------

static inline size_t align256(size_t x) { return (x + 255) & ~(size_t)255; }

extern "C" void kernel_launch(void* const* d_in, const int* in_sizes, int n_in,
                              void* d_out, int out_size, void* d_ws, size_t ws_size,
                              hipStream_t stream) {
  const float* x     = (const float*)d_in[0];
  const float* PA    = (const float*)d_in[1];
  const float* alpha = (const float*)d_in[2];
  const float* wa    = (const float*)d_in[3];
  const float* ba    = (const float*)d_in[4];
  const float* wb    = (const float*)d_in[5];
  const float* bb    = (const float*)d_in[6];
  const float* w1    = (const float*)d_in[7];
  const float* b1    = (const float*)d_in[8];
  const float* w2    = (const float*)d_in[9];
  const float* b2    = (const float*)d_in[10];
  const float* wd    = (const float*)d_in[11];
  const float* bd    = (const float*)d_in[12];
  float* y = (float*)d_out;

  char* ws = (char*)d_ws;
  size_t off = 0;
  bf16* xT = (bf16*)(ws + off);  off += align256((size_t)N_ * T_ * V_ * C_ * sizeof(bf16));
  bf16* wah = (bf16*)(ws + off); off += align256((size_t)S_ * O_ * C_ * sizeof(bf16));
  bf16* wbh = (bf16*)(ws + off); off += align256((size_t)S_ * O_ * C_ * sizeof(bf16));
  bf16* wdh = (bf16*)(ws + off); off += align256((size_t)S_ * O_ * C_ * sizeof(bf16));
  float* se_in = (float*)(ws + off); off += align256((size_t)N_ * C_ * T_ * sizeof(float));
  float* se1   = (float*)(ws + off); off += align256((size_t)S_ * N_ * IN_ * T_ * sizeof(float));
  float* sev   = (float*)(ws + off); off += align256((size_t)S_ * N_ * T_ * sizeof(float));
  bf16* aeffT  = (bf16*)(ws + off);  off += align256((size_t)S_ * N_ * VP * VP * sizeof(bf16));
  (void)ws_size; (void)n_in; (void)in_sizes; (void)out_size;

  // prep
  {
    int n = N_ * T_ * V_;
    agcn_prep_xT<<<(n + 255) / 256, 256, 0, stream>>>(x, xT);
  }
  {
    int n = S_ * O_ * C_;
    agcn_prep_w<<<(n + 255) / 256, 256, 0, stream>>>(wa, wb, wd, wah, wbh, wdh);
  }
  {
    int n = N_ * C_ * T_;
    agcn_se_mean<<<(n + 255) / 256, 256, 0, stream>>>(x, se_in);
  }
  // SE chain
  {
    int n = S_ * N_ * IN_ * T_;
    agcn_se_conv1<<<(n + 255) / 256, 256, 0, stream>>>(se_in, w1, b1, se1);
  }
  {
    int n = S_ * N_ * T_;
    agcn_se_conv2<<<(n + 255) / 256, 256, 0, stream>>>(se1, w2, b2, sev);
  }
  // attention -> A_eff (transposed bf16)
  agcn_attn<<<S_ * N_, 128, 0, stream>>>(xT, wah, wbh, ba, bb, PA, alpha, aeffT);
  // fused conv_d + graph conv + SE gate + subset sum
  agcn_out<<<N_ * T_, 128, 0, stream>>>(xT, wdh, bd, aeffT, sev, y);
}
